// RSSM_15006615733147
// MI455X (gfx1250) — compile-verified, matrix operand order checked by reading, not run on hardware
//
#include <hip/hip_runtime.h>

// ---------------- problem dims ----------------
#define BATCH 32768
#define ADIM  6
#define EDIM  1024
#define HDIM  200
#define DDIM  200
#define SDIM  32

// ---------------- tiling ----------------
#define KP   224     // padded H/D contraction dim (multiple of 32, >= 200)
#define NT   13      // 16-wide N tiles covering 208 >= 200
#define BM   128     // batch rows per block (8 waves x 16 rows)
#define KC   64      // K chunk for streaming E-dim GEMMs
#define NTHREADS 256

#define BUFA (BM * KC)      // 8192 elems  : one activation chunk buffer
#define BUFW (96 * KP)      // 21504 elems : one weight stage buffer (max tile)

// ---- prepacked bf16 weight layout inside d_ws (element offsets) ----
#define PK_ENC   0                          // [208][1024]
#define PK_ACT   (PK_ENC + 208 * 1024)      // [208][32]
#define PK_GRU   (PK_ACT + 208 * 32)        // [13][96][224] gate tiles
#define PK_PRIOR (PK_GRU + 13 * 96 * 224)   // [208][224]
#define PK_POSTH (PK_PRIOR + 208 * 224)     // [208][224]  (W_post h-part)
#define PK_POSTE (PK_POSTH + 208 * 224)     // [208][1024] (W_post obs-part)
#define PK_HP    (PK_POSTE + 208 * 1024)    // [64][224]   (W_pm | W_ps)
#define PK_HQ    (PK_HP + 64 * 224)         // [64][224]   (W_qm | W_qs)
#define PK_TOTAL (PK_HQ + 64 * 224)         // 834048 elems = 1668096 bytes

// ---- optional CDNA5 async global->LDS path (guarded; falls back to b128 copies)
#define USE_ASYNC 0
#if defined(__has_builtin)
#if __has_builtin(__builtin_amdgcn_global_load_async_to_lds_b128) && \
    __has_builtin(__builtin_amdgcn_s_wait_asynccnt)
#undef USE_ASYNC
#define USE_ASYNC 1
#endif
#endif

typedef __attribute__((ext_vector_type(16))) __bf16 bf16x16;
typedef __attribute__((ext_vector_type(8)))  float  f32x8;
typedef int v4i __attribute__((__vector_size__(4 * sizeof(int))));
typedef __attribute__((address_space(1))) v4i av4i_g;   // global  v4i
typedef __attribute__((address_space(3))) v4i av4i_l;   // LDS     v4i

union FragU { unsigned int u[8]; bf16x16 v; };

__device__ __forceinline__ unsigned short f2bf(float f) {
  unsigned int u = __float_as_uint(f);
  u += 0x7FFFu + ((u >> 16) & 1u);        // round-to-nearest-even
  return (unsigned short)(u >> 16);
}
__device__ __forceinline__ unsigned int pack2bf(float a, float b) {
  return (unsigned int)f2bf(a) | ((unsigned int)f2bf(b) << 16);
}
__device__ __forceinline__ float sigm(float x) { return 1.0f / (1.0f + __expf(-x)); }
__device__ __forceinline__ float softplus_(float x) {
  return fmaxf(x, 0.0f) + log1pf(__expf(-fabsf(x)));
}
__device__ __forceinline__ void wait_async_() {
#if USE_ASYNC
  __builtin_amdgcn_s_wait_asynccnt(0);
#endif
}

// scalar fp32->bf16 staging (used only for tiny ADIM=6 tiles)
__device__ __forceinline__ void stage_tile(unsigned short* dst, int ldd,
                                           const float* src, int lsrc,
                                           int rbase, int cbase,
                                           int nrows, int ncols,
                                           int maxRow, int maxCol, int tid) {
  const int total = nrows * ncols;
  for (int i = tid; i < total; i += NTHREADS) {
    int rr = i / ncols;
    int cc = i - rr * ncols;
    int gr = rbase + rr, gc = cbase + cc;
    float v = 0.0f;
    if (gr < maxRow && gc < maxCol) v = src[(size_t)gr * lsrc + gc];
    dst[rr * ldd + cc] = f2bf(v);
  }
}

// vectorized fp32->bf16 staging; requires cbase/ncols/maxCol/lsrc % 4 == 0
__device__ __forceinline__ void stage_cvt4(unsigned short* dst, int ldd,
                                           const float* src, int lsrc,
                                           int rbase, int cbase,
                                           int nrows, int ncols,
                                           int maxRow, int maxCol, int tid) {
  const int cq = ncols >> 2;
  const int total = nrows * cq;
  for (int i = tid; i < total; i += NTHREADS) {
    int rr = i / cq;
    int cc = (i - rr * cq) << 2;
    int gr = rbase + rr, gc = cbase + cc;
    float4 v = make_float4(0.f, 0.f, 0.f, 0.f);
    if (gr < maxRow && gc < maxCol)
      v = *reinterpret_cast<const float4*>(src + (size_t)gr * lsrc + gc);
    unsigned int* p = reinterpret_cast<unsigned int*>(dst + rr * ldd + cc);
    p[0] = pack2bf(v.x, v.y);
    p[1] = pack2bf(v.z, v.w);
  }
}

// raw bf16 tile copy global->LDS (prepacked weights); 16B granules.
__device__ __forceinline__ void stage_cpy(unsigned short* dst, int ldd,
                                          const unsigned short* src, int lsrc,
                                          int nrows, int ncols, int tid) {
  const int cq = ncols >> 3;
  const int total = nrows * cq;
  for (int i = tid; i < total; i += NTHREADS) {
    int rr = i / cq;
    int c8 = (i - rr * cq) << 3;
    const unsigned short* s = src + (size_t)rr * lsrc + c8;
    unsigned short* d = dst + rr * ldd + c8;
#if USE_ASYNC
    __builtin_amdgcn_global_load_async_to_lds_b128(
        (av4i_g*)(void*)s, (av4i_l*)(void*)d, 0, 0);
#else
    *reinterpret_cast<uint4*>(d) = *reinterpret_cast<const uint4*>(s);
#endif
  }
}

// 16x32 bf16 fragment: lane r = row (A) / out-col (B); half-wave picks K halves.
__device__ __forceinline__ bf16x16 ldfrag(const unsigned short* base, int ldd,
                                          int row, int kbase, int lh) {
  const unsigned short* p = base + row * ldd + kbase + lh * 8;
  uint4 lo = *reinterpret_cast<const uint4*>(p);
  uint4 hi = *reinterpret_cast<const uint4*>(p + 16);
  FragU f;
  f.u[0] = lo.x; f.u[1] = lo.y; f.u[2] = lo.z; f.u[3] = lo.w;
  f.u[4] = hi.x; f.u[5] = hi.y; f.u[6] = hi.z; f.u[7] = hi.w;
  return f.v;
}

__device__ __forceinline__ f32x8 wmma_bf16(bf16x16 a, bf16x16 b, f32x8 c) {
  return __builtin_amdgcn_wmma_f32_16x16x32_bf16(false, a, false, b, (short)0, c,
                                                 false, false);
}

// ---------------- weight prepack kernels (fp32 -> padded bf16 in d_ws) ----------------
__global__ void pack_pad(unsigned short* dst, const float* src, int lsrc,
                         int nrows, int ncols, int maxRow, int maxCol,
                         int rOff, int cOff) {
  const int total = nrows * ncols;
  for (int i = blockIdx.x * blockDim.x + threadIdx.x; i < total;
       i += gridDim.x * blockDim.x) {
    int rr = i / ncols, cc = i - rr * ncols;
    int gr = rOff + rr, gc = cOff + cc;
    float v = 0.f;
    if (gr < maxRow && gc < maxCol) v = src[(size_t)gr * lsrc + gc];
    dst[i] = f2bf(v);
  }
}

__global__ void pack_gru(unsigned short* dst, const float* w_ih, const float* w_hh) {
  const int total = 13 * 96 * 224;
  for (int i = blockIdx.x * blockDim.x + threadIdx.x; i < total;
       i += gridDim.x * blockDim.x) {
    int nt  = i / (96 * 224);
    int rem = i - nt * (96 * 224);
    int rr  = rem / 224;
    int cc  = rem - rr * 224;
    int g6 = rr >> 4, rl = rr & 15;
    int gate = g6 % 3;
    const float* src = (g6 < 3) ? w_ih : w_hh;
    int srow = gate * DDIM + nt * 16 + rl;
    float v = 0.f;
    if (srow < gate * DDIM + DDIM && cc < DDIM) v = src[srow * DDIM + cc];
    dst[i] = f2bf(v);
  }
}

// ---------------- fused RSSM step ----------------
__global__ __launch_bounds__(NTHREADS)
void rssm_fused(const float* __restrict__ action, const float* __restrict__ obs,
                const float* __restrict__ prev_h, const float* __restrict__ eps,
                const float* __restrict__ w_ih, const float* __restrict__ b_ih,
                const float* __restrict__ w_hh, const float* __restrict__ b_hh,
                const float* __restrict__ W_act, const float* __restrict__ b_act,
                const float* __restrict__ W_enc, const float* __restrict__ b_enc,
                const float* __restrict__ W_prior, const float* __restrict__ b_prior,
                const float* __restrict__ W_pm, const float* __restrict__ b_pm,
                const float* __restrict__ W_ps, const float* __restrict__ b_ps,
                const float* __restrict__ W_post, const float* __restrict__ b_post,
                const float* __restrict__ W_qm, const float* __restrict__ b_qm,
                const float* __restrict__ W_qs, const float* __restrict__ b_qs,
                const unsigned short* __restrict__ wpk, int pre,
                float* __restrict__ out) {
  extern __shared__ unsigned short smem[];
  unsigned short* sX  = smem;                 // [BM][KP] x   (later: prior)
  unsigned short* sHp = sX  + BM * KP;        // [BM][KP] h_prev (later: post)
  unsigned short* sHn = sHp + BM * KP;        // [BM][KP] new h
  unsigned short* sA  = sHn + BM * KP;        // 2 x [BM][KC] activation buffers
  unsigned short* sW  = sA  + 2 * BUFA;       // 2 x BUFW weight buffers

  const int tid  = threadIdx.x;
  const int wave = tid >> 5;
  const int lane = tid & 31;
  const int r    = lane & 15;
  const int lh   = lane >> 4;
  const int m0   = blockIdx.x * BM;
  const int rowA = wave * 16 + r;

  const size_t OH   = 0;
  const size_t OZ   = (size_t)BATCH * HDIM;
  const size_t OPM  = OZ   + (size_t)BATCH * SDIM;
  const size_t OPS  = OPM  + (size_t)BATCH * SDIM;
  const size_t OPRM = OPS  + (size_t)BATCH * SDIM;
  const size_t OPRS = OPRM + (size_t)BATCH * SDIM;

  const f32x8 vzero = {0.f, 0.f, 0.f, 0.f, 0.f, 0.f, 0.f, 0.f};

  // ---- zero pad columns of activation tiles; stage prev_h ----
  {
    unsigned int* z = (unsigned int*)smem;
    for (int i = tid; i < 3 * BM * KP / 2; i += NTHREADS) z[i] = 0u;
  }
  __syncthreads();
  stage_cvt4(sHp, KP, prev_h, DDIM, m0, 0, BM, KP, BATCH, DDIM, tid);

  // =========== Phase ENC: x = actW + relu(obs @ W_enc^T + b_enc) + b_act ===========
  f32x8 acc[NT];
#pragma unroll
  for (int t = 0; t < NT; ++t) acc[t] = vzero;

  if (pre) stage_cpy(sW, KC, wpk + PK_ENC, 1024, 208, KC, tid);
  else     stage_cvt4(sW, KC, W_enc, EDIM, 0, 0, 208, KC, HDIM, EDIM, tid);
  stage_cvt4(sA, KC, obs, EDIM, m0, 0, BM, KC, BATCH, EDIM, tid);
  wait_async_();
  __syncthreads();
#pragma unroll 1
  for (int kc = 0; kc < EDIM / KC; ++kc) {
    const int cur = kc & 1, nxt = cur ^ 1;
    if (kc + 1 < EDIM / KC) {
      if (pre) stage_cpy(sW + nxt * BUFW, KC, wpk + PK_ENC + (kc + 1) * KC, 1024, 208, KC, tid);
      else     stage_cvt4(sW + nxt * BUFW, KC, W_enc, EDIM, 0, (kc + 1) * KC, 208, KC, HDIM, EDIM, tid);
      stage_cvt4(sA + nxt * BUFA, KC, obs, EDIM, m0, (kc + 1) * KC, BM, KC, BATCH, EDIM, tid);
    }
    const unsigned short* cW = sW + cur * BUFW;
    const unsigned short* cA = sA + cur * BUFA;
    bf16x16 a0 = ldfrag(cA, KC, rowA, 0,  lh);
    bf16x16 a1 = ldfrag(cA, KC, rowA, 32, lh);
#pragma unroll
    for (int nt = 0; nt < NT; ++nt) {
      acc[nt] = wmma_bf16(a0, ldfrag(cW, KC, nt * 16 + r, 0,  lh), acc[nt]);
      acc[nt] = wmma_bf16(a1, ldfrag(cW, KC, nt * 16 + r, 32, lh), acc[nt]);
    }
    wait_async_();
    __syncthreads();
  }
  // action embed (K padded 6->32) + epilogue into sX
  if (pre) stage_cpy(sW, 32, wpk + PK_ACT, 32, 208, 32, tid);
  else     stage_tile(sW, 32, W_act, ADIM, 0, 0, 208, 32, HDIM, ADIM, tid);
  stage_tile(sA, 32, action, ADIM, m0, 0, BM, 32, BATCH, ADIM, tid);
  wait_async_();
  __syncthreads();
  {
    bf16x16 a0 = ldfrag(sA, 32, rowA, 0, lh);
#pragma unroll
    for (int nt = 0; nt < NT; ++nt) {
      f32x8 av = wmma_bf16(a0, ldfrag(sW, 32, nt * 16 + r, 0, lh), vzero);
      int n = nt * 16 + r;
      bool nv = n < HDIM;
      float ba = nv ? b_act[n] : 0.f;
      float be = nv ? b_enc[n] : 0.f;
#pragma unroll
      for (int j = 0; j < 8; ++j) {
        int rowL = wave * 16 + j + lh * 8;
        float e  = acc[nt][j] + be;
        float xv = av[j] + ba + (e > 0.f ? e : 0.f);
        sX[rowL * KP + n] = f2bf(nv ? xv : 0.f);
      }
    }
  }
  __syncthreads();

  // =========== Phase GRU (double-buffered gate-weight tiles) ===========
  if (pre) stage_cpy(sW, KP, wpk + PK_GRU, 224, 96, 224, tid);
  else {
#pragma unroll
    for (int g = 0; g < 3; ++g) {
      stage_cvt4(sW + (g * 16) * KP,       KP, w_ih, DDIM, g * DDIM, 0, 16, KP, g * DDIM + DDIM, DDIM, tid);
      stage_cvt4(sW + ((g + 3) * 16) * KP, KP, w_hh, DDIM, g * DDIM, 0, 16, KP, g * DDIM + DDIM, DDIM, tid);
    }
  }
  wait_async_();
  __syncthreads();
#pragma unroll 1
  for (int nt = 0; nt < NT; ++nt) {
    const int cur = nt & 1, nxt = cur ^ 1;
    if (nt + 1 < NT) {
      if (pre) stage_cpy(sW + nxt * BUFW, KP, wpk + PK_GRU + (size_t)(nt + 1) * 96 * 224, 224, 96, 224, tid);
      else {
#pragma unroll
        for (int g = 0; g < 3; ++g) {
          stage_cvt4(sW + nxt * BUFW + (g * 16) * KP,       KP, w_ih, DDIM,
                     g * DDIM + (nt + 1) * 16, 0, 16, KP, g * DDIM + DDIM, DDIM, tid);
          stage_cvt4(sW + nxt * BUFW + ((g + 3) * 16) * KP, KP, w_hh, DDIM,
                     g * DDIM + (nt + 1) * 16, 0, 16, KP, g * DDIM + DDIM, DDIM, tid);
        }
      }
    }
    const unsigned short* cW = sW + cur * BUFW;
    f32x8 air = vzero, aiz = vzero, ain = vzero;
    f32x8 ahr = vzero, ahz = vzero, ahn = vzero;
#pragma unroll
    for (int kk = 0; kk < KP / 32; ++kk) {
      bf16x16 ax = ldfrag(sX,  KP, rowA, kk * 32, lh);
      bf16x16 ah = ldfrag(sHp, KP, rowA, kk * 32, lh);
      air = wmma_bf16(ax, ldfrag(cW, KP,  0 + r, kk * 32, lh), air);
      aiz = wmma_bf16(ax, ldfrag(cW, KP, 16 + r, kk * 32, lh), aiz);
      ain = wmma_bf16(ax, ldfrag(cW, KP, 32 + r, kk * 32, lh), ain);
      ahr = wmma_bf16(ah, ldfrag(cW, KP, 48 + r, kk * 32, lh), ahr);
      ahz = wmma_bf16(ah, ldfrag(cW, KP, 64 + r, kk * 32, lh), ahz);
      ahn = wmma_bf16(ah, ldfrag(cW, KP, 80 + r, kk * 32, lh), ahn);
    }
    int n = nt * 16 + r;
    bool nv = n < DDIM;
    float bir = nv ? b_ih[n] : 0.f, biz = nv ? b_ih[DDIM + n] : 0.f,
          bin = nv ? b_ih[2 * DDIM + n] : 0.f;
    float bhr = nv ? b_hh[n] : 0.f, bhz = nv ? b_hh[DDIM + n] : 0.f,
          bhn = nv ? b_hh[2 * DDIM + n] : 0.f;
#pragma unroll
    for (int j = 0; j < 8; ++j) {
      int rowL = wave * 16 + j + lh * 8;
      float rg = sigm(air[j] + ahr[j] + bir + bhr);
      float zg = sigm(aiz[j] + ahz[j] + biz + bhz);
      float ng = tanhf(ain[j] + bin + rg * (ahn[j] + bhn));
      float hp = nv ? prev_h[(size_t)(m0 + rowL) * DDIM + n] : 0.f;
      float hv = (1.f - zg) * ng + zg * hp;
      if (nv) out[OH + (size_t)(m0 + rowL) * DDIM + n] = hv;
      sHn[rowL * KP + n] = f2bf(nv ? hv : 0.f);
    }
    wait_async_();
    __syncthreads();
  }

  // =========== Phase PRIOR: prior = relu(h @ W_prior^T + b) -> sX ===========
  if (pre) stage_cpy(sW, KP, wpk + PK_PRIOR, 224, 16, 224, tid);
  else     stage_cvt4(sW, KP, W_prior, DDIM, 0, 0, 16, KP, HDIM, DDIM, tid);
  wait_async_();
  __syncthreads();
#pragma unroll 1
  for (int nt = 0; nt < NT; ++nt) {
    const int cur = nt & 1, nxt = cur ^ 1;
    if (nt + 1 < NT) {
      if (pre) stage_cpy(sW + nxt * BUFW, KP, wpk + PK_PRIOR + (nt + 1) * 16 * 224, 224, 16, 224, tid);
      else     stage_cvt4(sW + nxt * BUFW, KP, W_prior, DDIM, (nt + 1) * 16, 0, 16, KP, HDIM, DDIM, tid);
    }
    const unsigned short* cW = sW + cur * BUFW;
    f32x8 ap = vzero;
#pragma unroll
    for (int kk = 0; kk < KP / 32; ++kk)
      ap = wmma_bf16(ldfrag(sHn, KP, rowA, kk * 32, lh),
                     ldfrag(cW, KP, r, kk * 32, lh), ap);
    int n = nt * 16 + r;
    bool nv = n < HDIM;
    float bp = nv ? b_prior[n] : 0.f;
#pragma unroll
    for (int j = 0; j < 8; ++j) {
      int rowL = wave * 16 + j + lh * 8;
      float v = ap[j] + bp; v = v > 0.f ? v : 0.f;
      sX[rowL * KP + n] = f2bf(nv ? v : 0.f);
    }
    wait_async_();
    __syncthreads();
  }

  // =========== Phase POST: post = relu([h,obs] @ W_post^T + b) -> sHp ===========
  f32x8 accp[NT];
#pragma unroll
  for (int t = 0; t < NT; ++t) accp[t] = vzero;

#pragma unroll 1
  for (int kc = 0; kc < 4; ++kc) {          // h-part: K = 224 (3x64 + 1x32)
    const int kcb = kc * 64;
    const int kw  = (kc == 3) ? 32 : 64;
    if (pre) stage_cpy(sW, KC, wpk + PK_POSTH + kcb, 224, 208, kw, tid);
    else     stage_cvt4(sW, KC, W_post, DDIM + EDIM, 0, kcb, 208, kw, HDIM, DDIM, tid);
    wait_async_();
    __syncthreads();
    for (int kk = 0; kk < kw / 32; ++kk) {
      bf16x16 a = ldfrag(sHn, KP, rowA, kcb + kk * 32, lh);
#pragma unroll
      for (int nt = 0; nt < NT; ++nt)
        accp[nt] = wmma_bf16(a, ldfrag(sW, KC, nt * 16 + r, kk * 32, lh), accp[nt]);
    }
    __syncthreads();
  }
  // obs-part: K = 1024, double-buffered
  if (pre) stage_cpy(sW, KC, wpk + PK_POSTE, 1024, 208, KC, tid);
  else     stage_cvt4(sW, KC, W_post, DDIM + EDIM, 0, DDIM, 208, KC, HDIM, DDIM + EDIM, tid);
  stage_cvt4(sA, KC, obs, EDIM, m0, 0, BM, KC, BATCH, EDIM, tid);
  wait_async_();
  __syncthreads();
#pragma unroll 1
  for (int kc = 0; kc < EDIM / KC; ++kc) {
    const int cur = kc & 1, nxt = cur ^ 1;
    if (kc + 1 < EDIM / KC) {
      if (pre) stage_cpy(sW + nxt * BUFW, KC, wpk + PK_POSTE + (kc + 1) * KC, 1024, 208, KC, tid);
      else     stage_cvt4(sW + nxt * BUFW, KC, W_post, DDIM + EDIM, 0, DDIM + (kc + 1) * KC,
                          208, KC, HDIM, DDIM + EDIM, tid);
      stage_cvt4(sA + nxt * BUFA, KC, obs, EDIM, m0, (kc + 1) * KC, BM, KC, BATCH, EDIM, tid);
    }
    const unsigned short* cW = sW + cur * BUFW;
    const unsigned short* cA = sA + cur * BUFA;
    bf16x16 a0 = ldfrag(cA, KC, rowA, 0,  lh);
    bf16x16 a1 = ldfrag(cA, KC, rowA, 32, lh);
#pragma unroll
    for (int nt = 0; nt < NT; ++nt) {
      accp[nt] = wmma_bf16(a0, ldfrag(cW, KC, nt * 16 + r, 0,  lh), accp[nt]);
      accp[nt] = wmma_bf16(a1, ldfrag(cW, KC, nt * 16 + r, 32, lh), accp[nt]);
    }
    wait_async_();
    __syncthreads();
  }
#pragma unroll
  for (int nt = 0; nt < NT; ++nt) {
    int n = nt * 16 + r;
    bool nv = n < HDIM;
    float bp = nv ? b_post[n] : 0.f;
#pragma unroll
    for (int j = 0; j < 8; ++j) {
      int rowL = wave * 16 + j + lh * 8;
      float v = accp[nt][j] + bp; v = v > 0.f ? v : 0.f;
      sHp[rowL * KP + n] = f2bf(nv ? v : 0.f);
    }
  }
  __syncthreads();

  // =========== Phase HEADS ===========
  if (pre) stage_cpy(sW, KP, wpk + PK_HP, 224, 64, 224, tid);
  else {
    stage_cvt4(sW,           KP, W_pm, HDIM, 0, 0, 32, KP, SDIM, HDIM, tid);
    stage_cvt4(sW + 32 * KP, KP, W_ps, HDIM, 0, 0, 32, KP, SDIM, HDIM, tid);
  }
  wait_async_();
  __syncthreads();
#pragma unroll
  for (int nt = 0; nt < 2; ++nt) {
    f32x8 am = vzero, as = vzero;
#pragma unroll
    for (int kk = 0; kk < KP / 32; ++kk) {
      bf16x16 a = ldfrag(sX, KP, rowA, kk * 32, lh);
      am = wmma_bf16(a, ldfrag(sW, KP,      nt * 16 + r, kk * 32, lh), am);
      as = wmma_bf16(a, ldfrag(sW, KP, 32 + nt * 16 + r, kk * 32, lh), as);
    }
    int n = nt * 16 + r;
    float bm = b_pm[n], bs = b_ps[n];
#pragma unroll
    for (int j = 0; j < 8; ++j) {
      int rowL = wave * 16 + j + lh * 8;
      size_t g = (size_t)(m0 + rowL) * SDIM + n;
      out[OPRM + g] = am[j] + bm;
      out[OPRS + g] = softplus_(as[j] + bs) + 0.1f;
    }
  }
  __syncthreads();
  if (pre) stage_cpy(sW, KP, wpk + PK_HQ, 224, 64, 224, tid);
  else {
    stage_cvt4(sW,           KP, W_qm, HDIM, 0, 0, 32, KP, SDIM, HDIM, tid);
    stage_cvt4(sW + 32 * KP, KP, W_qs, HDIM, 0, 0, 32, KP, SDIM, HDIM, tid);
  }
  wait_async_();
  __syncthreads();
#pragma unroll
  for (int nt = 0; nt < 2; ++nt) {
    f32x8 am = vzero, as = vzero;
#pragma unroll
    for (int kk = 0; kk < KP / 32; ++kk) {
      bf16x16 a = ldfrag(sHp, KP, rowA, kk * 32, lh);
      am = wmma_bf16(a, ldfrag(sW, KP,      nt * 16 + r, kk * 32, lh), am);
      as = wmma_bf16(a, ldfrag(sW, KP, 32 + nt * 16 + r, kk * 32, lh), as);
    }
    int n = nt * 16 + r;
    float bm = b_qm[n], bs = b_qs[n];
#pragma unroll
    for (int j = 0; j < 8; ++j) {
      int rowL = wave * 16 + j + lh * 8;
      size_t g = (size_t)(m0 + rowL) * SDIM + n;
      float pm = am[j] + bm;
      float ps = softplus_(as[j] + bs) + 0.1f;
      float ev = eps[g];
      out[OPM + g] = pm;
      out[OPS + g] = ps;
      out[OZ  + g] = pm + ps * ev;
    }
  }
}

extern "C" void kernel_launch(void* const* d_in, const int* in_sizes, int n_in,
                              void* d_out, int out_size, void* d_ws, size_t ws_size,
                              hipStream_t stream) {
  (void)in_sizes; (void)n_in; (void)out_size;
  const float* action  = (const float*)d_in[0];
  const float* obs     = (const float*)d_in[1];
  const float* prev_h  = (const float*)d_in[2];
  const float* eps     = (const float*)d_in[3];
  const float* w_ih    = (const float*)d_in[4];
  const float* b_ih    = (const float*)d_in[5];
  const float* w_hh    = (const float*)d_in[6];
  const float* b_hh    = (const float*)d_in[7];
  const float* W_act   = (const float*)d_in[8];
  const float* b_act   = (const float*)d_in[9];
  const float* W_enc   = (const float*)d_in[10];
  const float* b_enc   = (const float*)d_in[11];
  const float* W_prior = (const float*)d_in[12];
  const float* b_prior = (const float*)d_in[13];
  const float* W_pm    = (const float*)d_in[14];
  const float* b_pm    = (const float*)d_in[15];
  const float* W_ps    = (const float*)d_in[16];
  const float* b_ps    = (const float*)d_in[17];
  const float* W_post  = (const float*)d_in[18];
  const float* b_post  = (const float*)d_in[19];
  const float* W_qm    = (const float*)d_in[20];
  const float* b_qm    = (const float*)d_in[21];
  const float* W_qs    = (const float*)d_in[22];
  const float* b_qs    = (const float*)d_in[23];
  float* out = (float*)d_out;

  const int pre = (ws_size >= (size_t)PK_TOTAL * sizeof(unsigned short)) ? 1 : 0;
  unsigned short* wpk = (unsigned short*)d_ws;

  if (pre) {
    dim3 pb(256);
    pack_pad<<<dim3(832), pb, 0, stream>>>(wpk + PK_ENC,   W_enc,   1024, 208, 1024, 200, 1024, 0, 0);
    pack_pad<<<dim3(26),  pb, 0, stream>>>(wpk + PK_ACT,   W_act,   6,    208, 32,   200, 6,    0, 0);
    pack_gru<<<dim3(1092), pb, 0, stream>>>(wpk + PK_GRU, w_ih, w_hh);
    pack_pad<<<dim3(182), pb, 0, stream>>>(wpk + PK_PRIOR, W_prior, 200,  208, 224,  200, 200,  0, 0);
    pack_pad<<<dim3(182), pb, 0, stream>>>(wpk + PK_POSTH, W_post,  1224, 208, 224,  200, 200,  0, 0);
    pack_pad<<<dim3(832), pb, 0, stream>>>(wpk + PK_POSTE, W_post,  1224, 208, 1024, 200, 1224, 0, 200);
    pack_pad<<<dim3(28),  pb, 0, stream>>>(wpk + PK_HP,            W_pm, 200, 32, 224, 32, 200, 0, 0);
    pack_pad<<<dim3(28),  pb, 0, stream>>>(wpk + PK_HP + 32 * 224, W_ps, 200, 32, 224, 32, 200, 0, 0);
    pack_pad<<<dim3(28),  pb, 0, stream>>>(wpk + PK_HQ,            W_qm, 200, 32, 224, 32, 200, 0, 0);
    pack_pad<<<dim3(28),  pb, 0, stream>>>(wpk + PK_HQ + 32 * 224, W_qs, 200, 32, 224, 32, 200, 0, 0);
  }

  const size_t smem =
      (size_t)(3 * BM * KP + 2 * BUFA + 2 * BUFW) * sizeof(unsigned short);
  (void)hipFuncSetAttribute((const void*)rssm_fused,
                            hipFuncAttributeMaxDynamicSharedMemorySize, (int)smem);
  dim3 grid(BATCH / BM), block(NTHREADS);
  rssm_fused<<<grid, block, smem, stream>>>(
      action, obs, prev_h, eps, w_ih, b_ih, w_hh, b_hh,
      W_act, b_act, W_enc, b_enc, W_prior, b_prior,
      W_pm, b_pm, W_ps, b_ps, W_post, b_post,
      W_qm, b_qm, W_qs, b_qs, wpk, pre, out);
}
